// MBConvBlockMoE_38543036514533
// MI455X (gfx1250) — compile-verified, matrix operand order checked by reading, not run on hardware
//
#include <hip/hip_runtime.h>
#include <hip/hip_bf16.h>
#include <math.h>

// ---- problem dims (match reference) ----
#define Bn    64
#define CIN   96
#define Hd    28
#define Wd    28
#define HW    (Hd * Wd)        // 784
#define NE    4
#define HID   576
#define KSZ   5
#define PADV  2
#define RED   24
#define RHID  24
#define COUT  96
#define EPSV  1e-3f

typedef __attribute__((ext_vector_type(16))) __bf16 v16bf;
typedef __attribute__((ext_vector_type(8)))  __bf16 v8bf;
typedef __attribute__((ext_vector_type(8)))  float  v8f;
typedef int v4i __attribute__((vector_size(16)));

typedef __attribute__((address_space(1))) v4i* gas_v4i_ptr;
typedef __attribute__((address_space(3))) v4i* las_v4i_ptr;

__device__ __forceinline__ float dsigmoid(float x) { return 1.0f / (1.0f + __expf(-x)); }
__device__ __forceinline__ float dsilu(float x)    { return x * dsigmoid(x); }

// ---------------- async global->LDS copy (CDNA5 ASYNCcnt path) ----------------
#if __has_builtin(__builtin_amdgcn_global_load_async_to_lds_b128)
#define HAVE_ASYNC_LDS 1
#endif

// copy 16 bytes global -> LDS
__device__ __forceinline__ void cp16(const __bf16* g, __bf16* l) {
#ifdef HAVE_ASYNC_LDS
    __builtin_amdgcn_global_load_async_to_lds_b128(
        (gas_v4i_ptr)(size_t)g,
        (las_v4i_ptr)(unsigned)(size_t)l,
        0, 0);
#else
    *(v8bf*)l = *(const v8bf*)g;
#endif
}

__device__ __forceinline__ void async_wait0() {
#ifdef HAVE_ASYNC_LDS
#if __has_builtin(__builtin_amdgcn_s_wait_asynccnt)
    __builtin_amdgcn_s_wait_asynccnt(0);
#else
    asm volatile("s_wait_asynccnt 0x0" ::: "memory");
#endif
#endif
}

// Load one 16x32 bf16 WMMA fragment for this lane from a row-major,
// K-contiguous buffer (global).  Lane layout (ISA 7.12.2, 16-bit A 16x32):
// two contiguous 8-half runs at k = kk + hi*8 and k = kk + 16 + hi*8.
__device__ __forceinline__ v16bf ld_frag(const __bf16* __restrict__ rowptr,
                                         int kk, int hi) {
    const v8bf c0 = *(const v8bf*)(rowptr + kk + hi * 8);
    const v8bf c1 = *(const v8bf*)(rowptr + kk + 16 + hi * 8);
    v16bf r;
#pragma unroll
    for (int i = 0; i < 8; ++i) { r[i] = c0[i]; r[8 + i] = c1[i]; }
    return r;
}

// same, from an LDS panel row of 32 halves
__device__ __forceinline__ v16bf ld_frag_lds(const __bf16* rowptr, int hi) {
    const v8bf c0 = *(const v8bf*)(rowptr + hi * 8);
    const v8bf c1 = *(const v8bf*)(rowptr + 16 + hi * 8);
    v16bf r;
#pragma unroll
    for (int i = 0; i < 8; ++i) { r[i] = c0[i]; r[8 + i] = c1[i]; }
    return r;
}

// -------------------------------------------------------------------------
// 2M x NT-N register-blocked bf16 WMMA GEMM core with async double-buffered
// LDS staging of the shared B panel (B panel is contiguous: rows x KD halves).
//   Arow  : per-wave A rows (global, row-major KD)
//   Bpanel: block-shared B rows (global, row-major KD, contiguous)
//   ldsB  : 2 * NT*16*32 halves
// All waves of the block call this (workgroup-uniform barriers).
// -------------------------------------------------------------------------
template<int NT, int KD>
__device__ __forceinline__ void mm_block_lds(const __bf16* __restrict__ Arow,
                                             const __bf16* __restrict__ Bpanel,
                                             __bf16* ldsB,
                                             int lr, int hi, int tid, int nthr,
                                             v8f (&acc)[2][NT]) {
    const int BUF = NT * 16 * 32;      // halves per buffer
    const int NCH = NT * 64;           // 16-byte chunks per buffer (4 per row)

    // prologue: stage k-step 0 into buffer 0
    for (int ci = tid; ci < NCH; ci += nthr) {
        int row = ci >> 2, off = (ci & 3) * 8;
        cp16(Bpanel + (size_t)row * KD + off, ldsB + row * 32 + off);
    }
    async_wait0();
    __syncthreads();

    int it = 0;
    for (int kk = 0; kk < KD; kk += 32, ++it) {
        int cur = it & 1;
        // stage next k-step into the other buffer (overwrites the buffer
        // consumed in the previous iteration -- safe after last barrier)
        if (kk + 32 < KD) {
            __bf16* dst = ldsB + (cur ^ 1) * BUF;
            for (int ci = tid; ci < NCH; ci += nthr) {
                int row = ci >> 2, off = (ci & 3) * 8;
                cp16(Bpanel + (size_t)row * KD + kk + 32 + off, dst + row * 32 + off);
            }
        }
        // A fragments direct from global (tiny, L2-hot) + prefetch
        v16bf a0 = ld_frag(Arow + (size_t)lr * KD, kk, hi);
        v16bf a1 = ld_frag(Arow + (size_t)(16 + lr) * KD, kk, hi);
        if (kk + 32 < KD)
            __builtin_prefetch(Arow + (size_t)lr * KD + kk + 32, 0, 1);

        const __bf16* bbuf = ldsB + cur * BUF;
        v16bf bfr[NT];
#pragma unroll
        for (int j = 0; j < NT; ++j)
            bfr[j] = ld_frag_lds(bbuf + (j * 16 + lr) * 32, hi);
#pragma unroll
        for (int j = 0; j < NT; ++j) {
            acc[0][j] = __builtin_amdgcn_wmma_f32_16x16x32_bf16(
                false, a0, false, bfr[j], (short)0, acc[0][j], false, false);
            acc[1][j] = __builtin_amdgcn_wmma_f32_16x16x32_bf16(
                false, a1, false, bfr[j], (short)0, acc[1][j], false, false);
        }
        async_wait0();      // next buffer resident (this wave's copies)
        __syncthreads();    // ... and everyone else's; readers done with cur
    }
}

// -------------------------------------------------------------------------
// K0a: exp_w fp32 -> bf16
// K0b: x [b, c, p] fp32 -> xt [b, p, c] bf16  (K-contiguous B operand)
// -------------------------------------------------------------------------
__global__ void k0a_cvtw(const float* __restrict__ w, __bf16* __restrict__ wb, int n) {
    int i = blockIdx.x * blockDim.x + threadIdx.x;
    if (i < n) wb[i] = (__bf16)w[i];
}

__global__ void k0b_xt(const float* __restrict__ x, __bf16* __restrict__ xt) {
    int i = blockIdx.x * blockDim.x + threadIdx.x;
    if (i >= Bn * CIN * HW) return;
    int b = i / (CIN * HW);
    int r = i % (CIN * HW);
    int c = r / HW;
    int p = r % HW;
    xt[((size_t)b * HW + p) * CIN + c] = (__bf16)x[i];
}

// -------------------------------------------------------------------------
// K1: routing  (global avg pool -> MLP -> softmax) -> rw[B, E]
// -------------------------------------------------------------------------
__global__ void k1_routing(const float* __restrict__ x,
                           const float* __restrict__ r_w1, const float* __restrict__ r_b1,
                           const float* __restrict__ r_w2, const float* __restrict__ r_b2,
                           float* __restrict__ rw) {
    int b   = blockIdx.x;
    int tid = threadIdx.x;
    __shared__ float pool[CIN];
    __shared__ float hdn[RHID];
    __shared__ float logits[NE];
    if (tid < CIN) {
        const float* xp = x + ((size_t)b * CIN + tid) * HW;
        float s = 0.f;
        for (int i = 0; i < HW; ++i) s += xp[i];
        pool[tid] = s * (1.0f / HW);
    }
    __syncthreads();
    if (tid < RHID) {
        float s = r_b1[tid];
        for (int c = 0; c < CIN; ++c) s += pool[c] * r_w1[tid * CIN + c];
        hdn[tid] = fmaxf(s, 0.f);
    }
    __syncthreads();
    if (tid < NE) {
        float s = r_b2[tid];
        for (int j = 0; j < RHID; ++j) s += hdn[j] * r_w2[tid * RHID + j];
        logits[tid] = s;
    }
    __syncthreads();
    if (tid == 0) {
        float mx = logits[0];
        for (int e = 1; e < NE; ++e) mx = fmaxf(mx, logits[e]);
        float den = 0.f, ex[NE];
        for (int e = 0; e < NE; ++e) { ex[e] = __expf(logits[e] - mx); den += ex[e]; }
        float inv = 1.0f / den;
        for (int e = 0; e < NE; ++e) rw[b * NE + e] = ex[e] * inv;
    }
}

// -------------------------------------------------------------------------
// K2: expand GEMM  Y[b][576,784] = exp_wb[576,96] * xt[b]^T
//     6 waves/block share one 64-row B panel via async LDS double buffer.
//     block -> (b, g4, tmb); wave w handles m-group tm = tmb*6 + w
// -------------------------------------------------------------------------
template<int NT>
__device__ __forceinline__ void k2_body(const __bf16* Arow, const __bf16* Bpanel,
                                        __bf16* ldsB, int lr, int hi, int tid,
                                        int b, int mb0, int nb0,
                                        const float* g, const float* bb,
                                        const float* m, const float* v,
                                        __bf16* mid1) {
    v8f acc[2][NT] = {};
    mm_block_lds<NT, CIN>(Arow, Bpanel, ldsB, lr, hi, tid, 192, acc);
#pragma unroll
    for (int im = 0; im < 2; ++im)
#pragma unroll
        for (int j = 0; j < NT; ++j)
#pragma unroll
            for (int r = 0; r < 8; ++r) {
                int mm = mb0 + im * 16 + hi * 8 + r;
                int nn = nb0 + j * 16 + lr;
                float inv = g[mm] * rsqrtf(v[mm] + EPSV);
                float y = (acc[im][j][r] - m[mm]) * inv + bb[mm];
                mid1[((size_t)b * HID + mm) * HW + nn] = (__bf16)dsilu(y);
            }
}

__global__ __launch_bounds__(192)
void k2_expand(const __bf16* __restrict__ xt,
               const __bf16* __restrict__ wb,
               const float* __restrict__ g, const float* __restrict__ bb,
               const float* __restrict__ m, const float* __restrict__ v,
               __bf16* __restrict__ mid1) {
    __shared__ __bf16 ldsB[2 * 4 * 16 * 32];              // 8 KB
    int bx = blockIdx.x;
    int tmb = bx % 3; bx /= 3;
    int g4  = bx % 13;
    int b   = bx / 13;
    int tid = threadIdx.x;
    int wid = tid >> 5, lane = tid & 31;
    int lr = lane & 15, hi = lane >> 4;
    int tm  = tmb * 6 + wid;                              // 0..17
    int mb0 = tm * 32, nb0 = g4 * 64;
    const __bf16* Arow   = wb + (size_t)mb0 * CIN;
    const __bf16* Bpanel = xt + ((size_t)b * HW + nb0) * CIN;
    if (g4 != 12)
        k2_body<4>(Arow, Bpanel, ldsB, lr, hi, tid, b, mb0, nb0, g, bb, m, v, mid1);
    else
        k2_body<1>(Arow, Bpanel, ldsB, lr, hi, tid, b, mb0, nb0, g, bb, m, v, mid1);
}

// -------------------------------------------------------------------------
// K3: MoE depthwise 5x5 + BN2 + SiLU; writes mid2t [b, p, c] (transposed,
//     K-contiguous for K5's B operand) + channel sums for SE
// -------------------------------------------------------------------------
__global__ void k3_dwconv(const __bf16* __restrict__ mid1,
                          const float* __restrict__ rw,
                          const float* __restrict__ dw_w,
                          const float* __restrict__ g, const float* __restrict__ bb,
                          const float* __restrict__ m, const float* __restrict__ v,
                          __bf16* __restrict__ mid2t,
                          float* __restrict__ s_sum) {
    int b = blockIdx.x / HID;
    int c = blockIdx.x % HID;
    int tid = threadIdx.x;             // 256
    __shared__ float tile[32][32];
    __shared__ float kern[KSZ * KSZ];
    __shared__ float red[256];

    if (tid < KSZ * KSZ) {
        float s = 0.f;
        for (int e = 0; e < NE; ++e)
            s += rw[b * NE + e] * dw_w[((size_t)e * HID + c) * (KSZ * KSZ) + tid];
        kern[tid] = s;
    }
    const __bf16* plane = mid1 + ((size_t)b * HID + c) * HW;
    for (int i = tid; i < 32 * 32; i += 256) {
        int r  = (i >> 5) - PADV;
        int cc = (i & 31) - PADV;
        float val = 0.f;
        if (r >= 0 && r < Hd && cc >= 0 && cc < Wd) val = (float)plane[r * Wd + cc];
        tile[i >> 5][i & 31] = val;
    }
    __syncthreads();

    float inv = g[c] * rsqrtf(v[c] + EPSV);
    float lsum = 0.f;
    __bf16* obase = mid2t + (size_t)b * HW * HID + c;
    for (int p = tid; p < HW; p += 256) {
        int r = p / Wd, cc = p % Wd;
        float acc = 0.f;
#pragma unroll
        for (int ky = 0; ky < KSZ; ++ky)
#pragma unroll
            for (int kx = 0; kx < KSZ; ++kx)
                acc += kern[ky * KSZ + kx] * tile[r + ky][cc + kx];
        float y = (acc - m[c]) * inv + bb[c];
        y = dsilu(y);
        obase[(size_t)p * HID] = (__bf16)y;   // transposed store (L2 combines)
        lsum += y;
    }
    red[tid] = lsum;
    __syncthreads();
    for (int s = 128; s > 0; s >>= 1) {
        if (tid < s) red[tid] += red[tid + s];
        __syncthreads();
    }
    if (tid == 0) s_sum[b * HID + c] = red[0];
}

// -------------------------------------------------------------------------
// K4: squeeze-excitation MLP -> sc[B, HID]
// -------------------------------------------------------------------------
__global__ void k4_se(const float* __restrict__ s_sum,
                      const float* __restrict__ w1, const float* __restrict__ b1,
                      const float* __restrict__ w2, const float* __restrict__ b2,
                      float* __restrict__ sc) {
    int b = blockIdx.x;
    int tid = threadIdx.x;             // 256
    __shared__ float z[RED];
    if (tid < RED) {
        float s = b1[tid];
        for (int c = 0; c < HID; ++c)
            s += (s_sum[b * HID + c] * (1.0f / HW)) * w1[tid * HID + c];
        z[tid] = dsilu(s);
    }
    __syncthreads();
    for (int c = tid; c < HID; c += 256) {
        float s = b2[c];
        for (int j = 0; j < RED; ++j) s += z[j] * w2[c * RED + j];
        sc[b * HID + c] = dsigmoid(s);
    }
}

// -------------------------------------------------------------------------
// K4b: per-sample pointwise weights, SE scale folded into K columns:
//      wpwb[b, m, k] = bf16( (sum_e rw[b,e]*pw_w[e,m,k]) * sc[b,k] )
// -------------------------------------------------------------------------
__global__ void k4b_aggpw(const float* __restrict__ rw,
                          const float* __restrict__ pw_w,
                          const float* __restrict__ sc,
                          __bf16* __restrict__ wpwb) {
    int b = blockIdx.x;
    float r0 = rw[b * NE + 0], r1 = rw[b * NE + 1];
    float r2 = rw[b * NE + 2], r3 = rw[b * NE + 3];
    const size_t N = (size_t)COUT * HID;
    __bf16* o = wpwb + (size_t)b * N;
    const float* scb = sc + (size_t)b * HID;
    for (size_t i = threadIdx.x; i < N; i += blockDim.x) {
        int k = (int)(i % HID);
        float w = r0 * pw_w[i] + r1 * pw_w[N + i] + r2 * pw_w[2 * N + i] + r3 * pw_w[3 * N + i];
        o[i] = (__bf16)(w * scb[k]);
    }
}

// -------------------------------------------------------------------------
// K5: pointwise GEMM  out[b] = wpwb[b][96,576] * mid2t[b]^T  (+ BN3 + residual)
//     3 waves/block (all of COUT) share one 64-row B panel via async LDS.
// -------------------------------------------------------------------------
template<int NT>
__device__ __forceinline__ void k5_body(const __bf16* Arow, const __bf16* Bpanel,
                                        __bf16* ldsB, int lr, int hi, int tid,
                                        int b, int mb0, int nb0,
                                        const float* g, const float* bb,
                                        const float* m, const float* v,
                                        const float* x, float* out) {
    v8f acc[2][NT] = {};
    mm_block_lds<NT, HID>(Arow, Bpanel, ldsB, lr, hi, tid, 96, acc);
#pragma unroll
    for (int im = 0; im < 2; ++im)
#pragma unroll
        for (int j = 0; j < NT; ++j)
#pragma unroll
            for (int r = 0; r < 8; ++r) {
                int mm = mb0 + im * 16 + hi * 8 + r;
                int nn = nb0 + j * 16 + lr;
                float inv = g[mm] * rsqrtf(v[mm] + EPSV);
                float y = (acc[im][j][r] - m[mm]) * inv + bb[mm];
                size_t idx = ((size_t)b * COUT + mm) * HW + nn;
                out[idx] = x[idx] + y;   // residual
            }
}

__global__ __launch_bounds__(96)
void k5_pointwise(const __bf16* __restrict__ mid2t,
                  const __bf16* __restrict__ wpwb,
                  const float* __restrict__ x,
                  const float* __restrict__ g, const float* __restrict__ bb,
                  const float* __restrict__ m, const float* __restrict__ v,
                  float* __restrict__ out) {
    __shared__ __bf16 ldsB[2 * 4 * 16 * 32];              // 8 KB
    int bx = blockIdx.x;
    int g4 = bx % 13;
    int b  = bx / 13;
    int tid = threadIdx.x;
    int wid = tid >> 5, lane = tid & 31;
    int lr = lane & 15, hi = lane >> 4;
    int mb0 = wid * 32, nb0 = g4 * 64;                    // wid = 0..2
    const __bf16* Arow   = wpwb + (size_t)b * COUT * HID + (size_t)mb0 * HID;
    const __bf16* Bpanel = mid2t + ((size_t)b * HW + nb0) * HID;
    if (g4 != 12)
        k5_body<4>(Arow, Bpanel, ldsB, lr, hi, tid, b, mb0, nb0, g, bb, m, v, x, out);
    else
        k5_body<1>(Arow, Bpanel, ldsB, lr, hi, tid, b, mb0, nb0, g, bb, m, v, x, out);
}

// -------------------------------------------------------------------------
extern "C" void kernel_launch(void* const* d_in, const int* in_sizes, int n_in,
                              void* d_out, int out_size, void* d_ws, size_t ws_size,
                              hipStream_t stream) {
    const float* x     = (const float*)d_in[0];
    const float* r_w1  = (const float*)d_in[1];
    const float* r_b1  = (const float*)d_in[2];
    const float* r_w2  = (const float*)d_in[3];
    const float* r_b2  = (const float*)d_in[4];
    const float* exp_w = (const float*)d_in[5];
    const float* bn1_g = (const float*)d_in[6];
    const float* bn1_b = (const float*)d_in[7];
    const float* bn1_m = (const float*)d_in[8];
    const float* bn1_v = (const float*)d_in[9];
    const float* dw_w  = (const float*)d_in[10];
    const float* bn2_g = (const float*)d_in[11];
    const float* bn2_b = (const float*)d_in[12];
    const float* bn2_m = (const float*)d_in[13];
    const float* bn2_v = (const float*)d_in[14];
    const float* se_w1 = (const float*)d_in[15];
    const float* se_b1 = (const float*)d_in[16];
    const float* se_w2 = (const float*)d_in[17];
    const float* se_b2 = (const float*)d_in[18];
    const float* pw_w  = (const float*)d_in[19];
    const float* bn3_g = (const float*)d_in[20];
    const float* bn3_b = (const float*)d_in[21];
    const float* bn3_m = (const float*)d_in[22];
    const float* bn3_v = (const float*)d_in[23];
    float* out = (float*)d_out;

    // workspace carving (bytes, 1KB aligned)
    char*  ws  = (char*)d_ws;
    size_t off = 0;
    auto carve = [&](size_t bytes) { size_t o = off; off = (off + bytes + 1023) & ~size_t(1023); return o; };
    float*  rw    = (float*)(ws + carve(sizeof(float) * Bn * NE));
    float*  s_sum = (float*)(ws + carve(sizeof(float) * Bn * HID));
    float*  scv   = (float*)(ws + carve(sizeof(float) * Bn * HID));
    __bf16* wexpb = (__bf16*)(ws + carve(sizeof(__bf16) * HID * CIN));
    __bf16* xt    = (__bf16*)(ws + carve(sizeof(__bf16) * (size_t)Bn * HW * CIN));
    __bf16* wpwb  = (__bf16*)(ws + carve(sizeof(__bf16) * (size_t)Bn * COUT * HID));
    __bf16* mid1  = (__bf16*)(ws + carve(sizeof(__bf16) * (size_t)Bn * HID * HW));
    __bf16* mid2t = (__bf16*)(ws + carve(sizeof(__bf16) * (size_t)Bn * HW * HID));
    (void)ws_size; (void)in_sizes; (void)n_in; (void)out_size;

    // 0) bf16 operand prep
    k0a_cvtw<<<(HID * CIN + 255) / 256, 256, 0, stream>>>(exp_w, wexpb, HID * CIN);
    k0b_xt<<<(Bn * CIN * HW + 255) / 256, 256, 0, stream>>>(x, xt);

    // 1) routing
    k1_routing<<<Bn, 128, 0, stream>>>(x, r_w1, r_b1, r_w2, r_b2, rw);

    // 2) expand GEMM (bf16 WMMA, async-LDS staged B) + BN1 + SiLU
    k2_expand<<<Bn * 13 * 3, 192, 0, stream>>>(xt, wexpb, bn1_g, bn1_b, bn1_m, bn1_v, mid1);

    // 3) MoE depthwise + BN2 + SiLU (transposed output + channel sums)
    k3_dwconv<<<Bn * HID, 256, 0, stream>>>(mid1, rw, dw_w, bn2_g, bn2_b, bn2_m, bn2_v,
                                            mid2t, s_sum);

    // 4) SE MLP -> scale; 4b) aggregate pointwise weights (bf16, sc folded)
    k4_se<<<Bn, 256, 0, stream>>>(s_sum, se_w1, se_b1, se_w2, se_b2, scv);
    k4b_aggpw<<<Bn, 256, 0, stream>>>(rw, pw_w, scv, wpwb);

    // 5) pointwise GEMM (bf16 WMMA, async-LDS staged B) + BN3 + residual
    k5_pointwise<<<Bn * 13, 96, 0, stream>>>(mid2t, wpwb, x, bn3_g, bn3_b, bn3_m, bn3_v, out);
}
